// GCN_fusion1_91036126806360
// MI455X (gfx1250) — compile-verified, matrix operand order checked by reading, not run on hardware
//
#include <hip/hip_runtime.h>
#include <math.h>

typedef __attribute__((ext_vector_type(16))) _Float16 v16h;
typedef __attribute__((ext_vector_type(8)))  _Float16 v8h;
typedef __attribute__((ext_vector_type(4)))  _Float16 v4h;
typedef __attribute__((ext_vector_type(8)))  float    v8f;

#define SELU_LAMBDA 1.0507009873554805f
#define SELU_ALPHA  1.6732632423543772f

__device__ __forceinline__ float selu_f(float x) {
    return SELU_LAMBDA * (x > 0.f ? x : SELU_ALPHA * (__expf(x) - 1.f));
}

// ---------------------------------------------------------------------------
// Zero fill (graph-capture safe; used instead of hipMemsetAsync)
// ---------------------------------------------------------------------------
__global__ void fill_zero(float* __restrict__ p, long n) {
    long i = (long)blockIdx.x * blockDim.x + threadIdx.x;
    long stride = (long)gridDim.x * blockDim.x;
    for (; i < n; i += stride) p[i] = 0.f;
}

// ---------------------------------------------------------------------------
// f32 -> f16 convert, 4 elements per thread iteration (n must be multiple of 4)
// ---------------------------------------------------------------------------
__global__ void cvt_f16(const float* __restrict__ in, _Float16* __restrict__ out,
                        long n4) {
    long i = (long)blockIdx.x * blockDim.x + threadIdx.x;
    const long stride = (long)gridDim.x * blockDim.x;
    for (; i < n4; i += stride) {
        const float4 v = ((const float4*)in)[i];
        v4h h;
        h[0] = (_Float16)v.x; h[1] = (_Float16)v.y;
        h[2] = (_Float16)v.z; h[3] = (_Float16)v.w;
        ((v4h*)out)[i] = h;
    }
}

// ---------------------------------------------------------------------------
// W[K,N] f32 row-major -> Wt[N,K] f16 row-major (tiny matrices)
// ---------------------------------------------------------------------------
__global__ void transpose_cvt_f16(const float* __restrict__ W,
                                  _Float16* __restrict__ Wt, int K, int N) {
    int i = blockIdx.x * blockDim.x + threadIdx.x;
    if (i >= K * N) return;
    const int k = i / N;
    const int n = i - k * N;
    Wt[(long)n * K + k] = (_Float16)W[i];
}

// ---------------------------------------------------------------------------
// WMMA GEMM: C[M,N] = A[M,K] @ Bt[N,K]^T ; A, Bt are f16 row-major,
// C fp32. One wave32 per 16x16 C tile, 8 waves per block.
// Per the CDNA5 16-bit fragment layout, lane (idx, half) holds contiguous
// K-octets at half*8 and 16+half*8 -> pure global_load_b128 fragment loads.
// K unrolled by 64 -> 2 v_wmma per iteration. Requires M%16==0, N%16==0,
// K%64==0 (true here: M=50000, K=256/512, N=512/128).
// ---------------------------------------------------------------------------
__global__ __launch_bounds__(256)
void wmma_gemm_f16t(const _Float16* __restrict__ A,
                    const _Float16* __restrict__ Bt,
                    float* __restrict__ C, int M, int N, int K) {
    const int wave = threadIdx.x >> 5;
    const int lane = threadIdx.x & 31;
    const int tiles_n = N >> 4;
    const int tile = blockIdx.x * 8 + wave;
    const int total = (M >> 4) * tiles_n;
    if (tile >= total) return;              // wave-uniform: EXEC stays all-1s

    const int tm = tile / tiles_n;
    const int tn = tile - tm * tiles_n;
    const int idx  = lane & 15;             // A: row-in-tile; B/C: col-in-tile
    const int half = lane >> 4;

    const _Float16* Ar = A  + (long)((tm << 4) + idx) * K + (half << 3);
    const _Float16* Br = Bt + (long)((tn << 4) + idx) * K + (half << 3);

    v8f acc = {};
    for (int k0 = 0; k0 < K; k0 += 64) {
        const v8h aL0 = *(const v8h*)(Ar + k0);
        const v8h aH0 = *(const v8h*)(Ar + k0 + 16);
        const v8h bL0 = *(const v8h*)(Br + k0);
        const v8h bH0 = *(const v8h*)(Br + k0 + 16);
        const v8h aL1 = *(const v8h*)(Ar + k0 + 32);
        const v8h aH1 = *(const v8h*)(Ar + k0 + 48);
        const v8h bL1 = *(const v8h*)(Br + k0 + 32);
        const v8h bH1 = *(const v8h*)(Br + k0 + 48);

        const v16h a0 = __builtin_shufflevector(aL0, aH0, 0,1,2,3,4,5,6,7,
                                                8,9,10,11,12,13,14,15);
        const v16h b0 = __builtin_shufflevector(bL0, bH0, 0,1,2,3,4,5,6,7,
                                                8,9,10,11,12,13,14,15);
        acc = __builtin_amdgcn_wmma_f32_16x16x32_f16(
            false, a0, false, b0, (short)0, acc, false, false);

        const v16h a1 = __builtin_shufflevector(aL1, aH1, 0,1,2,3,4,5,6,7,
                                                8,9,10,11,12,13,14,15);
        const v16h b1 = __builtin_shufflevector(bL1, bH1, 0,1,2,3,4,5,6,7,
                                                8,9,10,11,12,13,14,15);
        acc = __builtin_amdgcn_wmma_f32_16x16x32_f16(
            false, a1, false, b1, (short)0, acc, false, false);
    }

    // C layout: VGPR r -> row (tm*16 + half*8 + r), col (tn*16 + idx)
    float* Crow = C + (long)((tm << 4) + (half << 3)) * N + (tn << 4) + idx;
#pragma unroll
    for (int r = 0; r < 8; ++r) Crow[(long)r * N] = acc[r];
}

// ---------------------------------------------------------------------------
// SpMM via atomics: out[row[e], :] += val[e] * dense[col[e], :]
// One work item = (edge, 4 features): float4 gather + 4 f32 atomic adds.
// dense (<=102.4MB) fits in the 192MB L2, so gathers/atomics stay on-chip.
// ---------------------------------------------------------------------------
__global__ void spmm_atomic(const int* __restrict__ rows,
                            const int* __restrict__ cols,
                            const float* __restrict__ vals,
                            const float* __restrict__ dense,
                            float* __restrict__ out, int E, int F) {
    const int fq = F >> 2;
    const long total = (long)E * fq;
    long w = (long)blockIdx.x * blockDim.x + threadIdx.x;
    const long stride = (long)gridDim.x * blockDim.x;
    for (; w < total; w += stride) {
        const int e  = (int)(w / fq);
        const int f4 = (int)(w - (long)e * fq) << 2;
        const int r = rows[e];
        const int c = cols[e];
        const float v = vals[e];
        const float4 d = *(const float4*)(dense + (long)c * F + f4);
        float* o = out + (long)r * F + f4;
        atomicAdd(o + 0, v * d.x);
        atomicAdd(o + 1, v * d.y);
        atomicAdd(o + 2, v * d.z);
        atomicAdd(o + 3, v * d.w);
    }
}

// ---------------------------------------------------------------------------
// h[i] = selu(h[i] + bias[i % F])
// ---------------------------------------------------------------------------
__global__ void bias_selu(float* __restrict__ h, const float* __restrict__ bias,
                          long n, int F) {
    long i = (long)blockIdx.x * blockDim.x + threadIdx.x;
    const long stride = (long)gridDim.x * blockDim.x;
    for (; i < n; i += stride) {
        const int f = (int)(i % F);
        h[i] = selu_f(h[i] + bias[f]);
    }
}

// ---------------------------------------------------------------------------
// Mean-pool partial: each block sums 64 rows locally per column (128 cols),
// then one atomicAdd per (block, col).
// ---------------------------------------------------------------------------
__global__ __launch_bounds__(128)
void pool_partial(const float* __restrict__ h, float* __restrict__ acc, int Nrows) {
    const int c = threadIdx.x;                   // 0..127
    const int r0 = blockIdx.x * 64;
    const int rend = (r0 + 64 < Nrows) ? (r0 + 64) : Nrows;
    float s = 0.f;
    for (int r = r0; r < rend; ++r) s += h[(long)r * 128 + c];
    atomicAdd(&acc[c], s);
}

// ---------------------------------------------------------------------------
// Head: g = selu(mean) ; concat sub_fea ; logits = g @ Wf + bf ;
// log_softmax ; l1 = mean(|Wf|). Single block of 128 threads.
// ---------------------------------------------------------------------------
__global__ __launch_bounds__(128)
void head_kernel(const float* __restrict__ acc, const float* __restrict__ sub_fea,
                 const float* __restrict__ Wf, const float* __restrict__ bf,
                 float* __restrict__ out, int Nrows) {
    __shared__ float g[192];
    __shared__ float red[128];
    const int t = threadIdx.x;                   // 0..127

    g[t] = selu_f(acc[t] / (float)Nrows);        // pooled nclass part
    if (t < 64) g[128 + t] = sub_fea[t];         // sub_fea part
    __syncthreads();

    // logits[t] = bf[t] + sum_i g[i] * Wf[i*128 + t]
    float logit = bf[t];
#pragma unroll 4
    for (int i = 0; i < 192; ++i) logit += g[i] * Wf[(long)i * 128 + t];

    // max reduce
    red[t] = logit; __syncthreads();
    for (int s = 64; s > 0; s >>= 1) {
        if (t < s) red[t] = fmaxf(red[t], red[t + s]);
        __syncthreads();
    }
    const float mx = red[0]; __syncthreads();

    // sum-exp reduce
    red[t] = __expf(logit - mx); __syncthreads();
    for (int s = 64; s > 0; s >>= 1) {
        if (t < s) red[t] += red[t + s];
        __syncthreads();
    }
    const float lse = mx + logf(red[0]); __syncthreads();
    out[t] = logit - lse;

    // l1 = mean(|Wf|) over 192*128 elements
    float s1 = 0.f;
    for (int i = t; i < 192 * 128; i += 128) s1 += fabsf(Wf[i]);
    red[t] = s1; __syncthreads();
    for (int s = 64; s > 0; s >>= 1) {
        if (t < s) red[t] += red[t + s];
        __syncthreads();
    }
    if (t == 0) out[128] = red[0] / (float)(192 * 128);
}

// ---------------------------------------------------------------------------
extern "C" void kernel_launch(void* const* d_in, const int* in_sizes, int n_in,
                              void* d_out, int out_size, void* d_ws, size_t ws_size,
                              hipStream_t stream) {
    (void)n_in; (void)out_size; (void)ws_size;

    const float* x   = (const float*)d_in[0];
    const int*   adj = (const int*)  d_in[1];
    const float* av  = (const float*)d_in[2];
    const float* sub = (const float*)d_in[3];
    const float* W1  = (const float*)d_in[4];
    const float* b1  = (const float*)d_in[5];
    const float* W2  = (const float*)d_in[6];
    const float* b2  = (const float*)d_in[7];
    const float* Wf  = (const float*)d_in[8];
    const float* bf  = (const float*)d_in[9];
    float* out = (float*)d_out;

    const int NF = 256, NH = 512, NC = 128;
    const int N = in_sizes[0] / NF;              // 50000 (multiple of 16)
    const int E = in_sizes[1] / 2;               // 1.6M
    const int* rows = adj;
    const int* cols = adj + E;

    // workspace layout (floats):
    //   bufA [N*NH] | bufB [N*NH] | a16 [N*NH halves = N*NH/2 floats]
    //   | wt16 [512*512 halves] | pooled [NC]
    float* bufA      = (float*)d_ws;
    float* bufB      = bufA + (long)N * NH;
    _Float16* a16    = (_Float16*)(bufB + (long)N * NH);
    _Float16* wt16   = a16 + (long)N * NH;
    float* pooled    = (float*)(wt16 + 512 * 512);

    // ---- layer 1: bufA = x @ W1  (M=N, N=NH, K=NF) ----
    cvt_f16<<<2048, 256, 0, stream>>>(x, a16, (long)N * NF / 4);
    transpose_cvt_f16<<<(NF * NH + 255) / 256, 256, 0, stream>>>(W1, wt16, NF, NH);
    {
        const int tiles = (N >> 4) * (NH >> 4);
        wmma_gemm_f16t<<<(tiles + 7) / 8, 256, 0, stream>>>(a16, wt16, bufA,
                                                            N, NH, NF);
    }
    fill_zero<<<2048, 256, 0, stream>>>(bufB, (long)N * NH);
    spmm_atomic<<<8192, 256, 0, stream>>>(rows, cols, av, bufA, bufB, E, NH);
    bias_selu<<<2048, 256, 0, stream>>>(bufB, b1, (long)N * NH, NH);

    // ---- layer 2: bufA = h1 @ W2  (M=N, N=NC, K=NH) ----
    cvt_f16<<<2048, 256, 0, stream>>>(bufB, a16, (long)N * NH / 4);
    transpose_cvt_f16<<<(NH * NC + 255) / 256, 256, 0, stream>>>(W2, wt16, NH, NC);
    {
        const int tiles = (N >> 4) * (NC >> 4);
        wmma_gemm_f16t<<<(tiles + 7) / 8, 256, 0, stream>>>(a16, wt16, bufA,
                                                            N, NC, NH);
    }
    fill_zero<<<2048, 256, 0, stream>>>(bufB, (long)N * NC);
    spmm_atomic<<<8192, 256, 0, stream>>>(rows, cols, av, bufA, bufB, E, NC);
    bias_selu<<<2048, 256, 0, stream>>>(bufB, b2, (long)N * NC, NC);

    // ---- pool + head ----
    fill_zero<<<1, 128, 0, stream>>>(pooled, NC);
    pool_partial<<<(N + 63) / 64, 128, 0, stream>>>(bufB, pooled, N);
    head_kernel<<<1, 128, 0, stream>>>(pooled, sub, Wf, bf, out, N);
}
